// GCN_31224412242835
// MI455X (gfx1250) — matrix operand order repaired and checked by reading, not compiled, not run on hardware
//
#include <hip/hip_runtime.h>

#define N_NODES 50000
#define N_PAD   50016                 // padded to multiple of 32 rows (no guards in GEMM)
#define N_EDGES 1600000
#define IN_DIM  1024
#define HID     2048
#define OUTC    3
#define BN_EPS  1e-5f
#define M_PAIRS (N_PAD / 32)          // 1563 wave macro-tiles (32 rows each)

typedef __attribute__((ext_vector_type(16))) __bf16 v16bf;
typedef __attribute__((ext_vector_type(8)))  __bf16 v8bf;
typedef __attribute__((ext_vector_type(8)))  float  v8f;

__device__ __forceinline__ __bf16 f2bf(float f) {
  unsigned u = __builtin_bit_cast(unsigned, f);
  u += 0x7fffu + ((u >> 16) & 1u);           // round-to-nearest-even
  unsigned short s = (unsigned short)(u >> 16);
  return __builtin_bit_cast(__bf16, s);
}

// ---------------- utility ----------------
__global__ void zero_f32_kernel(float* __restrict__ p, size_t n) {
  size_t i = (size_t)blockIdx.x * blockDim.x + threadIdx.x;
  if (i < n) p[i] = 0.0f;
}

// ---------------- degrees / norms ----------------
__global__ void degree_kernel(const int* __restrict__ src, const int* __restrict__ dst,
                              float* __restrict__ deg_out, float* __restrict__ deg_in) {
  int e = blockIdx.x * blockDim.x + threadIdx.x;
  if (e >= N_EDGES) return;
  unsafeAtomicAdd(&deg_out[src[e]], 1.0f);
  unsafeAtomicAdd(&deg_in[dst[e]], 1.0f);
}

__global__ void norm_kernel(const float* __restrict__ deg_out, const float* __restrict__ deg_in,
                            float* __restrict__ out_norm, float* __restrict__ in_norm) {
  int i = blockIdx.x * blockDim.x + threadIdx.x;
  if (i >= N_NODES) return;
  out_norm[i] = rsqrtf(fmaxf(deg_out[i], 1.0f));
  in_norm[i]  = rsqrtf(fmaxf(deg_in[i], 1.0f));
}

// ---------------- W1 -> bf16, fragment-major pack ----------------
// Tile (k0/32, n0/16): 32 lanes x 16 halves contiguous.
// lane l: col = n0 + (l&15); k = k0 + (h>>3)*16 + (l>>4)*8 + (h&7), h=0..15  (ISA 16-bit B 32x16 layout)
__global__ void pack_w1_kernel(const float* __restrict__ w1, __bf16* __restrict__ w1p) {
  int i = blockIdx.x * blockDim.x + threadIdx.x;
  if (i >= IN_DIM * HID) return;
  int within = i & 511;                       // element inside 32x16 tile
  int tIdx   = i >> 9;
  int l = within >> 4, h = within & 15;
  int kt = tIdx / (HID / 16), nt = tIdx % (HID / 16);
  int k   = kt * 32 + (h >> 3) * 16 + (l >> 4) * 8 + (h & 7);
  int col = nt * 16 + (l & 15);
  w1p[i] = f2bf(w1[(size_t)k * HID + col]);
}

// ---------------- layer-1 edge scatter with BOTH norms folded ----------------
// agg[dst] += in_feat[src] * (out_norm[src] * in_norm[dst])
__global__ void scatter1_kernel(const float* __restrict__ x, const int* __restrict__ src,
                                const int* __restrict__ dst, const float* __restrict__ out_norm,
                                const float* __restrict__ in_norm, float* __restrict__ agg) {
  const int e = blockIdx.x;
  const int s = src[e], d = dst[e];
  const float sc = out_norm[s] * in_norm[d];
  const float* xr = x + (size_t)s * IN_DIM;
  float* ar = agg + (size_t)d * IN_DIM;
#pragma unroll
  for (int i = 0; i < 4; ++i) {
    int c = threadIdx.x + i * 256;
    unsafeAtomicAdd(&ar[c], xr[c] * sc);
  }
}

// ---------------- agg f32 -> bf16 A-matrix (single cheap pass) ----------------
__global__ void cvt_a_kernel(const float* __restrict__ agg, __bf16* __restrict__ abf) {
  size_t i = (size_t)blockIdx.x * blockDim.x + threadIdx.x;
  if (i < (size_t)N_PAD * IN_DIM) abf[i] = f2bf(agg[i]);
}

// ---------------- GEMM1: H = relu(A @ W1 + b1), bf16 WMMA, f32 accum ----------------
// block = 256 thr = 8 waves; wave -> 32 rows x 64 cols = 2x4 WMMA tiles, 8 WMMA / k-step
__global__ void gemm1_kernel(const __bf16* __restrict__ abf, const __bf16* __restrict__ w1p,
                             const float* __restrict__ b1, float* __restrict__ hout) {
  const int lane  = threadIdx.x & 31;
  const int wave  = threadIdx.x >> 5;
  const int mPair = blockIdx.x * 8 + wave;
  if (mPair >= M_PAIRS) return;
  const int nBase = blockIdx.y * 64;
  const int rsel  = lane & 15;
  const int hi    = lane >> 4;
  const int row0  = mPair * 32 + rsel;        // first M-tile
  const int row1  = row0 + 16;                // second M-tile

  v8f acc[2][4] = {{v8f{}, v8f{}, v8f{}, v8f{}}, {v8f{}, v8f{}, v8f{}, v8f{}}};

  for (int k0 = 0; k0 < IN_DIM; k0 += 32) {
    // A fragments: halves 0..7 -> k0+hi*8+j ; 8..15 -> k0+16+hi*8+j (two 16B vector loads each)
    const __bf16* ap0 = abf + (size_t)row0 * IN_DIM + k0 + hi * 8;
    const __bf16* ap1 = abf + (size_t)row1 * IN_DIM + k0 + hi * 8;
    v8bf a0l = *(const v8bf*)ap0, a0h = *(const v8bf*)(ap0 + 16);
    v8bf a1l = *(const v8bf*)ap1, a1h = *(const v8bf*)(ap1 + 16);
    v16bf a0 = __builtin_shufflevector(a0l, a0h, 0,1,2,3,4,5,6,7,8,9,10,11,12,13,14,15);
    v16bf a1 = __builtin_shufflevector(a1l, a1h, 0,1,2,3,4,5,6,7,8,9,10,11,12,13,14,15);

    const __bf16* bbase =
        w1p + ((size_t)(k0 >> 5) * (HID / 16) + (nBase >> 4)) * 512 + lane * 16;
#pragma unroll
    for (int nt = 0; nt < 4; ++nt) {
      v16bf b = *(const v16bf*)(bbase + (size_t)nt * 512);   // one 32B fragment load
      acc[0][nt] = __builtin_amdgcn_wmma_f32_16x16x32_bf16(
          false, a0, false, b, (short)0, acc[0][nt], false, false);
      acc[1][nt] = __builtin_amdgcn_wmma_f32_16x16x32_bf16(
          false, a1, false, b, (short)0, acc[1][nt], false, false);
    }
  }

  // epilogue: bias + relu; C/D layout: vgpr r -> row hi*8+r, col rsel
#pragma unroll
  for (int m = 0; m < 2; ++m) {
#pragma unroll
    for (int nt = 0; nt < 4; ++nt) {
      const int col = nBase + nt * 16 + rsel;
      const float bias = b1[col];
#pragma unroll
      for (int r = 0; r < 8; ++r) {
        const int orow = mPair * 32 + m * 16 + hi * 8 + r;
        hout[(size_t)orow * HID + col] = fmaxf(acc[m][nt][r] + bias, 0.0f);
      }
    }
  }
}

// ---------------- BN column stats (sum, sumsq) ----------------
__global__ void colstats_kernel(const float* __restrict__ h,
                                float* __restrict__ colsum, float* __restrict__ colsq) {
  const int r0 = blockIdx.x * 256;
  const int r1 = min(r0 + 256, N_NODES);
  float s[8] = {0,0,0,0,0,0,0,0}, q[8] = {0,0,0,0,0,0,0,0};
  for (int r = r0; r < r1; ++r) {
    const float* hr = h + (size_t)r * HID;
#pragma unroll
    for (int i = 0; i < 8; ++i) {
      float v = hr[threadIdx.x + i * 256];
      s[i] += v; q[i] += v * v;
    }
  }
#pragma unroll
  for (int i = 0; i < 8; ++i) {
    unsafeAtomicAdd(&colsum[threadIdx.x + i * 256], s[i]);
    unsafeAtomicAdd(&colsq[threadIdx.x + i * 256], q[i]);
  }
}

__global__ void bn_finalize_kernel(const float* __restrict__ colsum, const float* __restrict__ colsq,
                                   float* __restrict__ meanv, float* __restrict__ invv) {
  int c = blockIdx.x * blockDim.x + threadIdx.x;
  if (c >= HID) return;
  float m = colsum[c] * (1.0f / N_NODES);
  float v = colsq[c] * (1.0f / N_NODES) - m * m;
  meanv[c] = m;
  invv[c]  = rsqrtf(v + BN_EPS);
}

// ---------------- fold BN into Wout ----------------
__global__ void wp_kernel(const float* __restrict__ meanv, const float* __restrict__ invv,
                          const float* __restrict__ wout, float* __restrict__ wp,
                          float* __restrict__ cj) {
  __shared__ float red[256];
  const int j = blockIdx.x;
  float p = 0.0f;
  for (int k = threadIdx.x; k < HID; k += 256) {
    float w = invv[k] * wout[k * OUTC + j];
    wp[k * OUTC + j] = w;
    p += meanv[k] * w;
  }
  red[threadIdx.x] = p;
  __syncthreads();
  for (int s = 128; s > 0; s >>= 1) {
    if (threadIdx.x < s) red[threadIdx.x] += red[threadIdx.x + s];
    __syncthreads();
  }
  if (threadIdx.x == 0) cj[j] = red[0];
}

// ---------------- project: t[i,:] = out_norm[i] * (H[i,:] @ wp - cj) ----------------
__global__ void fc2_kernel(const float* __restrict__ h, const float* __restrict__ wp,
                           const float* __restrict__ cj, const float* __restrict__ out_norm,
                           float* __restrict__ t) {
  __shared__ float swp[HID * OUTC];
  __shared__ float red[256 * 3];
  const int i = blockIdx.x;
  for (int idx = threadIdx.x; idx < HID * OUTC; idx += 256) swp[idx] = wp[idx];
  __syncthreads();
  float a0 = 0.f, a1 = 0.f, a2 = 0.f;
  const float* hr = h + (size_t)i * HID;
  for (int k = threadIdx.x; k < HID; k += 256) {
    float hv = hr[k];
    a0 += hv * swp[k * 3 + 0];
    a1 += hv * swp[k * 3 + 1];
    a2 += hv * swp[k * 3 + 2];
  }
  red[threadIdx.x] = a0; red[256 + threadIdx.x] = a1; red[512 + threadIdx.x] = a2;
  __syncthreads();
  for (int s = 128; s > 0; s >>= 1) {
    if (threadIdx.x < s) {
      red[threadIdx.x]       += red[threadIdx.x + s];
      red[256 + threadIdx.x] += red[256 + threadIdx.x + s];
      red[512 + threadIdx.x] += red[512 + threadIdx.x + s];
    }
    __syncthreads();
  }
  if (threadIdx.x == 0) {
    float on = out_norm[i];
    t[(size_t)i * 3 + 0] = on * (red[0]   - cj[0]);
    t[(size_t)i * 3 + 1] = on * (red[256] - cj[1]);
    t[(size_t)i * 3 + 2] = on * (red[512] - cj[2]);
  }
}

// ---------------- layer-2 edge scatter over 3 features ----------------
__global__ void scatter2_kernel(const int* __restrict__ src, const int* __restrict__ dst,
                                const float* __restrict__ t, float* __restrict__ acc2) {
  int e = blockIdx.x * blockDim.x + threadIdx.x;
  if (e >= N_EDGES) return;
  int s = src[e], d = dst[e];
  float v0 = t[(size_t)s * 3 + 0], v1 = t[(size_t)s * 3 + 1], v2 = t[(size_t)s * 3 + 2];
  unsafeAtomicAdd(&acc2[(size_t)d * 3 + 0], v0);
  unsafeAtomicAdd(&acc2[(size_t)d * 3 + 1], v1);
  unsafeAtomicAdd(&acc2[(size_t)d * 3 + 2], v2);
}

// ---------------- in_norm scale + bias + softmax ----------------
__global__ void softmax_kernel(const float* __restrict__ acc2, const float* __restrict__ in_norm,
                               const float* __restrict__ bout, float* __restrict__ out) {
  int i = blockIdx.x * blockDim.x + threadIdx.x;
  if (i >= N_NODES) return;
  float n  = in_norm[i];
  float v0 = acc2[(size_t)i * 3 + 0] * n + bout[0];
  float v1 = acc2[(size_t)i * 3 + 1] * n + bout[1];
  float v2 = acc2[(size_t)i * 3 + 2] * n + bout[2];
  float m  = fmaxf(v0, fmaxf(v1, v2));
  float e0 = __expf(v0 - m), e1 = __expf(v1 - m), e2 = __expf(v2 - m);
  float inv = 1.0f / (e0 + e1 + e2);
  out[(size_t)i * 3 + 0] = e0 * inv;
  out[(size_t)i * 3 + 1] = e1 * inv;
  out[(size_t)i * 3 + 2] = e2 * inv;
}

extern "C" void kernel_launch(void* const* d_in, const int* in_sizes, int n_in,
                              void* d_out, int out_size, void* d_ws, size_t ws_size,
                              hipStream_t stream) {
  const float* in_feat = (const float*)d_in[0];
  const int*   src     = (const int*)d_in[1];
  const int*   dst     = (const int*)d_in[2];
  const float* W1      = (const float*)d_in[3];
  const float* b1      = (const float*)d_in[4];
  const float* Wout    = (const float*)d_in[5];
  const float* bout    = (const float*)d_in[6];
  float* out = (float*)d_out;

  // ---- workspace layout (zeroed buffers packed first) ----
  char* base = (char*)d_ws;
  size_t off = 0;
  auto take = [&](size_t bytes) -> void* {
    void* p = base + off;
    off += (bytes + 255) & ~(size_t)255;
    return p;
  };
  float* deg_out = (float*)take((size_t)N_NODES * 4);
  float* deg_in  = (float*)take((size_t)N_NODES * 4);
  float* agg     = (float*)take((size_t)N_PAD * IN_DIM * 4);   // padded, pad rows stay 0
  float* colsum  = (float*)take((size_t)HID * 4);
  float* colsq   = (float*)take((size_t)HID * 4);
  float* acc2    = (float*)take((size_t)N_NODES * 3 * 4);
  const size_t zero_elems = off / 4;          // everything above must start at 0
  float*  out_norm = (float*)take((size_t)N_NODES * 4);
  float*  in_norm  = (float*)take((size_t)N_NODES * 4);
  __bf16* w1p      = (__bf16*)take((size_t)IN_DIM * HID * 2);  // fragment-packed
  __bf16* abf      = (__bf16*)take((size_t)N_PAD * IN_DIM * 2);
  float*  hbuf     = (float*)take((size_t)N_PAD * HID * 4);
  float*  meanv    = (float*)take((size_t)HID * 4);
  float*  invv     = (float*)take((size_t)HID * 4);
  float*  wp       = (float*)take((size_t)HID * 3 * 4);
  float*  cj       = (float*)take(16);
  float*  tbuf     = (float*)take((size_t)N_NODES * 3 * 4);
  (void)ws_size; (void)in_sizes; (void)n_in; (void)out_size;

  // 1. zero the accumulation region
  zero_f32_kernel<<<dim3((unsigned)((zero_elems + 255) / 256)), 256, 0, stream>>>((float*)base, zero_elems);
  // 2. degrees + norms
  degree_kernel<<<dim3((N_EDGES + 255) / 256), 256, 0, stream>>>(src, dst, deg_out, deg_in);
  norm_kernel<<<dim3((N_NODES + 255) / 256), 256, 0, stream>>>(deg_out, deg_in, out_norm, in_norm);
  // 3. W1 -> bf16 fragment-packed
  pack_w1_kernel<<<dim3((IN_DIM * HID + 255) / 256), 256, 0, stream>>>(W1, w1p);
  // 4. layer-1 edge aggregation, both norms folded (dominant: ~13 GB random traffic)
  scatter1_kernel<<<dim3(N_EDGES), 256, 0, stream>>>(in_feat, src, dst, out_norm, in_norm, agg);
  // 5. A-matrix to bf16 (one streaming pass)
  cvt_a_kernel<<<dim3((unsigned)(((size_t)N_PAD * IN_DIM + 255) / 256)), 256, 0, stream>>>(agg, abf);
  // 6. GEMM1 + bias + relu via bf16 WMMA (32x64 per wave, 8 WMMA / k-step)
  gemm1_kernel<<<dim3((M_PAIRS + 7) / 8, HID / 64), 256, 0, stream>>>(abf, w1p, b1, hbuf);
  // 7. BN batch stats
  colstats_kernel<<<dim3((N_NODES + 255) / 256), 256, 0, stream>>>(hbuf, colsum, colsq);
  bn_finalize_kernel<<<dim3(HID / 256), 256, 0, stream>>>(colsum, colsq, meanv, invv);
  // 8. fold BN into Wout
  wp_kernel<<<dim3(3), 256, 0, stream>>>(meanv, invv, Wout, wp, cj);
  // 9. project H to 3 cols before the scatter (commutes with segment_sum)
  fc2_kernel<<<dim3(N_NODES), 256, 0, stream>>>(hbuf, wp, cj, out_norm, tbuf);
  // 10. layer-2 edge scatter (3 floats/edge instead of 2048)
  scatter2_kernel<<<dim3((N_EDGES + 255) / 256), 256, 0, stream>>>(src, dst, tbuf, acc2);
  // 11. in_norm + bias + softmax
  softmax_kernel<<<dim3((N_NODES + 255) / 256), 256, 0, stream>>>(acc2, in_norm, bout, out);
}
